// MultiHeadedAttention_22101901705644
// MI455X (gfx1250) — matrix operand order repaired and checked by reading, not compiled, not run on hardware
//
#include <hip/hip_runtime.h>
#include <math.h>

#define DEV __device__ __forceinline__

typedef __attribute__((ext_vector_type(4)))  float  v4f;
typedef __attribute__((ext_vector_type(8)))  float  v8f;
typedef __attribute__((ext_vector_type(8)))  __bf16 v8bf;
typedef __attribute__((ext_vector_type(16))) __bf16 v16bf;

constexpr int B_ = 2, S_ = 2048, D_ = 1024, H_ = 16, DK_ = 64;

// fp32 -> bf16, round-to-nearest-even via integer path (compile-safe everywhere)
DEV __bf16 f2bf(float f) {
    union { float f; unsigned u; } in; in.f = f;
    unsigned r = in.u + 0x7FFFu + ((in.u >> 16) & 1u);
    union { unsigned short s; __bf16 b; } out; out.s = (unsigned short)(r >> 16);
    return out.b;
}

union Frag { v16bf v; v8bf h[2]; };

DEV void store_out(float*  p, float v) { *p = v; }
DEV void store_out(__bf16* p, float v) { *p = f2bf(v); }

// A-frag (16x32 bf16) from LDS: 16 contiguous K elems (K=16h..+16)
DEV Frag fragA_lds(const __bf16* p) {
    Frag f; f.h[0] = *(const v8bf*)p; f.h[1] = *(const v8bf*)(p + 8); return f;
}
// B-frag (32x16 bf16) from LDS: 16 contiguous K elems
DEV Frag fragB_lds(const __bf16* p) {
    Frag f; f.h[0] = *(const v8bf*)p; f.h[1] = *(const v8bf*)(p + 8); return f;
}

// cross-lane xor shuffle via ds_swizzle (group-of-32 mode: and=0x1F, or=0, xor=M)
template<int M>
DEV float swz_xor(float v) {
    return __int_as_float(__builtin_amdgcn_ds_swizzle(__float_as_int(v), (M << 10) | 0x1F));
}
DEV float grp16_max(float v) {
    v = fmaxf(v, swz_xor<1>(v)); v = fmaxf(v, swz_xor<2>(v));
    v = fmaxf(v, swz_xor<4>(v)); v = fmaxf(v, swz_xor<8>(v));
    return v;
}
DEV float grp16_sum(float v) {
    v += swz_xor<1>(v); v += swz_xor<2>(v); v += swz_xor<4>(v); v += swz_xor<8>(v);
    return v;
}

// CDNA5 async copy: global -> LDS, 16 bytes per lane, tracked by ASYNCcnt
DEV void async_g2l_b128(unsigned lds_off, const __bf16* gptr) {
    asm volatile("global_load_async_to_lds_b128 %0, %1, off"
                 :: "v"(lds_off), "v"(gptr) : "memory");
}
DEV void wait_async0() { asm volatile("s_wait_asynccnt 0" ::: "memory"); }
DEV void wait_ds0()    { asm volatile("s_wait_dscnt 0"    ::: "memory"); }

// ---------------------------------------------------------------------------
// One-shot fp32 -> bf16 staging (bandwidth-bound; keeps GEMM inner loops pure)
// ---------------------------------------------------------------------------
__global__ __launch_bounds__(256)
void conv_f32_bf16(const float* __restrict__ in, __bf16* __restrict__ out, int n) {
    int base = (blockIdx.x * 256 + threadIdx.x) * 8;
    if (base >= n) return;
    v4f a = *(const v4f*)(in + base), b = *(const v4f*)(in + base + 4);
    v8bf r;
    r[0] = f2bf(a[0]); r[1] = f2bf(a[1]); r[2] = f2bf(a[2]); r[3] = f2bf(a[3]);
    r[4] = f2bf(b[0]); r[5] = f2bf(b[1]); r[6] = f2bf(b[2]); r[7] = f2bf(b[3]);
    *(v8bf*)(out + base) = r;
}

// ---------------------------------------------------------------------------
// Y[M,N] = X[M,K] @ W[N,K]^T + bias — all-bf16 WMMA.
// Block 256 thr = 8 waves (2M x 4N), tile 64x128; wave = 32x32.
// K staged 64-wide into double-buffered LDS via global_load_async_to_lds_b128;
// fragments read with ds_load_b128; DMA of stage s+1 overlaps WMMA of stage s.
// LDS row stride 72 elems (144B) to spread banks; 16B aligned.
// ---------------------------------------------------------------------------
template<typename TOUT>
__global__ __launch_bounds__(256)
void gemm_bias_wmma(const __bf16* __restrict__ X, const __bf16* __restrict__ W,
                    const float* __restrict__ bias, TOUT* __restrict__ Y,
                    int M, int N, int K) {
    __shared__ __bf16 Alds[2][64][72];     // 64 rows x 64 K (+8 pad)
    __shared__ __bf16 Wlds[2][128][72];    // 128 rows x 64 K (+8 pad)

    const int tid  = threadIdx.x;
    const int lane = tid & 31;
    const int wave = tid >> 5;                   // 0..7
    const int wm   = wave & 1, wn = wave >> 1;   // 2 x 4 wave grid
    const int m16  = lane & 15, half = lane >> 4;
    const int row0 = blockIdx.y * 64  + wm * 32;
    const int col0 = blockIdx.x * 128 + wn * 32;

    const __bf16* Xg = X + (size_t)(blockIdx.y * 64)  * K;
    const __bf16* Wg = W + (size_t)(blockIdx.x * 128) * K;
    const unsigned abase[2] = { (unsigned)(size_t)&Alds[0][0][0], (unsigned)(size_t)&Alds[1][0][0] };
    const unsigned wbase[2] = { (unsigned)(size_t)&Wlds[0][0][0], (unsigned)(size_t)&Wlds[1][0][0] };

    // cooperative async stage of one 64-wide K slice: 512 A chunks + 1024 W chunks
    auto stage = [&](int k0, int buf) {
#pragma unroll
        for (int i = 0; i < 2; ++i) {            // A tile 64x64
            int c = tid + 256 * i, row = c >> 3, col8 = c & 7;
            async_g2l_b128(abase[buf] + (unsigned)(row * 144 + col8 * 16),
                           Xg + (size_t)row * K + k0 + col8 * 8);
        }
#pragma unroll
        for (int i = 0; i < 4; ++i) {            // W tile 128x64
            int c = tid + 256 * i, row = c >> 3, col8 = c & 7;
            async_g2l_b128(wbase[buf] + (unsigned)(row * 144 + col8 * 16),
                           Wg + (size_t)row * K + k0 + col8 * 8);
        }
    };

    v8f c00 = {}, c01 = {}, c10 = {}, c11 = {};
    const int nsteps = K / 64;
    stage(0, 0);
    for (int s = 0; s < nsteps; ++s) {
        wait_async0();
        __syncthreads();                         // stage s ready; prior reads done
        if (s + 1 < nsteps) stage((s + 1) * 64, (s + 1) & 1);
        const __bf16* ab = &Alds[s & 1][0][0];
        const __bf16* wb = &Wlds[s & 1][0][0];
#pragma unroll
        for (int ks = 0; ks < 64; ks += 32) {
            Frag a0 = fragA_lds(ab + (wm * 32 +      m16) * 72 + ks + 16 * half);
            Frag a1 = fragA_lds(ab + (wm * 32 + 16 + m16) * 72 + ks + 16 * half);
            Frag b0 = fragB_lds(wb + (wn * 32 +      m16) * 72 + ks + 16 * half);
            Frag b1 = fragB_lds(wb + (wn * 32 + 16 + m16) * 72 + ks + 16 * half);
            c00 = __builtin_amdgcn_wmma_f32_16x16x32_bf16(false, a0.v, false, b0.v, (short)0, c00, false, false);
            c01 = __builtin_amdgcn_wmma_f32_16x16x32_bf16(false, a0.v, false, b1.v, (short)0, c01, false, false);
            c10 = __builtin_amdgcn_wmma_f32_16x16x32_bf16(false, a1.v, false, b0.v, (short)0, c10, false, false);
            c11 = __builtin_amdgcn_wmma_f32_16x16x32_bf16(false, a1.v, false, b1.v, (short)0, c11, false, false);
        }
    }
    const int n0 = col0 + m16;
    const float bv0 = bias[n0], bv1 = bias[n0 + 16];
#pragma unroll
    for (int r = 0; r < 8; ++r) {
        int rowA = row0 + r + 8 * half;          // C layout: M = r + 8*(lane/16)
        int rowB = rowA + 16;
        store_out(Y + (size_t)rowA * N + n0,      c00[r] + bv0);
        store_out(Y + (size_t)rowA * N + n0 + 16, c01[r] + bv1);
        store_out(Y + (size_t)rowB * N + n0,      c10[r] + bv0);
        store_out(Y + (size_t)rowB * N + n0 + 16, c11[r] + bv1);
    }
}

// ---------------------------------------------------------------------------
// Causal flash attention. blockIdx.y = (b,h); 128 q-rows/block, 8 waves x 16.
// All waves iterate to the block causal limit (uniform trip count); each
// 32-key K and V chunk is async-staged into double-buffered LDS and shared by
// all 8 waves; per-wave compute is predicated on its own causal limit.
// ---------------------------------------------------------------------------
__global__ __launch_bounds__(256)
void attn_fwd(const __bf16* __restrict__ Q, const __bf16* __restrict__ Kb,
              const __bf16* __restrict__ V, __bf16* __restrict__ X) {
    __shared__ __bf16 Klds[2][32][80];           // staged K chunk (row = 160B)
    __shared__ __bf16 Vlds[2][32][80];           // staged V chunk
    __shared__ __bf16 Plds[8][16][40];           // per-wave P re-tiling buffer

    const int tid  = threadIdx.x;
    const int lane = tid & 31;
    const int wave = tid >> 5;                   // 0..7
    const int n    = lane & 15;
    const int half = lane >> 4;
    const int bh   = blockIdx.y;
    const int b    = bh / H_, h = bh % H_;
    const int q0b  = blockIdx.x * 128;
    const int q0   = q0b + wave * 16;

    const __bf16* qp = Q  + ((size_t)b * S_) * D_ + h * DK_;
    const __bf16* kp = Kb + ((size_t)b * S_) * D_ + h * DK_;
    const __bf16* vp = V  + ((size_t)b * S_) * D_ + h * DK_;

    const unsigned kbase[2] = { (unsigned)(size_t)&Klds[0][0][0], (unsigned)(size_t)&Klds[1][0][0] };
    const unsigned vbase[2] = { (unsigned)(size_t)&Vlds[0][0][0], (unsigned)(size_t)&Vlds[1][0][0] };

    // cooperative async stage of K/V rows [kt..kt+31] x 64 cols (2 x 256 chunks)
    auto stage = [&](int kt, int buf) {
        int row = tid >> 3, col8 = tid & 7;
        unsigned off = (unsigned)(row * 160 + col8 * 16);
        async_g2l_b128(kbase[buf] + off, kp + (size_t)(kt + row) * D_ + col8 * 8);
        async_g2l_b128(vbase[buf] + off, vp + (size_t)(kt + row) * D_ + col8 * 8);
    };

    // Q A-fragments: two 16x32 tiles covering d = 0..63
    Frag qf[2];
#pragma unroll
    for (int f = 0; f < 2; ++f) {
        const __bf16* p = qp + (size_t)(q0 + n) * D_ + 32 * f + 16 * half;
        qf[f].h[0] = *(const v8bf*)p;
        qf[f].h[1] = *(const v8bf*)(p + 8);
    }

    v8f o[4] = {};                               // 16x64 f32 accumulator
    float mrow[8], lrow[8];
#pragma unroll
    for (int r = 0; r < 8; ++r) { mrow[r] = -INFINITY; lrow[r] = 0.f; }

    const float scale = 0.125f;                  // 1/sqrt(DK)
    const int qmax  = q0 + 15;                   // this wave's causal limit
    const int ktend = q0b + 127;                 // block causal limit (uniform)

    stage(0, 0);
    for (int kt = 0, s = 0; kt <= ktend; kt += 32, ++s) {
        wait_async0();
        __syncthreads();                         // chunk s staged; prior reads done
        if (kt + 32 <= ktend) stage(kt + 32, (s + 1) & 1);
        if (kt > qmax) continue;                 // fully-masked for this wave

        const __bf16* kbuf = &Klds[s & 1][0][0];
        const __bf16* vbuf = &Vlds[s & 1][0][0];

        // ---- S = Q K^T: two 16-key subtiles, K-frags from LDS ---------
        Frag kb00 = fragB_lds(kbuf + (n     ) * 80 +      16 * half);
        Frag kb10 = fragB_lds(kbuf + (n + 16) * 80 +      16 * half);
        Frag kb01 = fragB_lds(kbuf + (n     ) * 80 + 32 + 16 * half);
        Frag kb11 = fragB_lds(kbuf + (n + 16) * 80 + 32 + 16 * half);
        v8f s0 = {}, s1 = {};
        s0 = __builtin_amdgcn_wmma_f32_16x16x32_bf16(false, qf[0].v, false, kb00.v, (short)0, s0, false, false);
        s0 = __builtin_amdgcn_wmma_f32_16x16x32_bf16(false, qf[1].v, false, kb01.v, (short)0, s0, false, false);
        s1 = __builtin_amdgcn_wmma_f32_16x16x32_bf16(false, qf[0].v, false, kb10.v, (short)0, s1, false, false);
        s1 = __builtin_amdgcn_wmma_f32_16x16x32_bf16(false, qf[1].v, false, kb11.v, (short)0, s1, false, false);

        // ---- scale + causal mask --------------------------------------
#pragma unroll
        for (int r = 0; r < 8; ++r) {
            int row = q0 + r + 8 * half;
            float a = s0[r] * scale, c = s1[r] * scale;
            if (kt + n > row)      a = -INFINITY;
            if (kt + 16 + n > row) c = -INFINITY;
            s0[r] = a; s1[r] = c;
        }
        wait_ds0();                              // WAR vs previous P loads
        // ---- online softmax + stage P to LDS --------------------------
#pragma unroll
        for (int r = 0; r < 8; ++r) {
            float mt   = grp16_max(fmaxf(s0[r], s1[r]));
            float mnew = fmaxf(mrow[r], mt);
            float alpha = __expf(mrow[r] - mnew);
            float p0 = __expf(s0[r] - mnew);
            float p1 = __expf(s1[r] - mnew);
            lrow[r] = lrow[r] * alpha + grp16_sum(p0 + p1);
            mrow[r] = mnew;
#pragma unroll
            for (int dt = 0; dt < 4; ++dt) o[dt][r] *= alpha;
            int rl = r + 8 * half;
            Plds[wave][rl][n]      = f2bf(p0);
            Plds[wave][rl][16 + n] = f2bf(p1);
        }
        wait_ds0();                              // RAW: stores -> loads
        // ---- reload P in A-fragment layout ----------------------------
        Frag pf;
        pf.h[0] = *(const v8bf*)&Plds[wave][n][16 * half];
        pf.h[1] = *(const v8bf*)&Plds[wave][n][16 * half + 8];
        // ---- O += P V, V B-frags gathered from LDS --------------------
#pragma unroll
        for (int dt = 0; dt < 4; ++dt) {
            union { v16bf v; __bf16 e[16]; } vb;
#pragma unroll
            for (int j = 0; j < 16; ++j)
                vb.e[j] = vbuf[(size_t)(16 * half + j) * 80 + 16 * dt + n];
            o[dt] = __builtin_amdgcn_wmma_f32_16x16x32_bf16(false, pf.v, false, vb.v, (short)0, o[dt], false, false);
        }
    }
    // ---- normalize + write context [B,S,D] as bf16 --------------------
#pragma unroll
    for (int r = 0; r < 8; ++r) {
        float inv = 1.0f / lrow[r];
        int row = q0 + r + 8 * half;
        __bf16* xp = X + ((size_t)b * S_ + row) * D_ + h * DK_;
#pragma unroll
        for (int dt = 0; dt < 4; ++dt)
            xp[16 * dt + n] = f2bf(o[dt][r] * inv);
    }
}

// ---------------------------------------------------------------------------
extern "C" void kernel_launch(void* const* d_in, const int* in_sizes, int n_in,
                              void* d_out, int out_size, void* d_ws, size_t ws_size,
                              hipStream_t stream) {
    (void)in_sizes; (void)n_in; (void)out_size; (void)ws_size;
    const float* query = (const float*)d_in[0];
    const float* key   = (const float*)d_in[1];
    const float* value = (const float*)d_in[2];
    /* d_in[3] = mask: causal tril, computed analytically in-kernel */
    const float* Wq = (const float*)d_in[4];  const float* bq = (const float*)d_in[5];
    const float* Wk = (const float*)d_in[6];  const float* bk = (const float*)d_in[7];
    const float* Wv = (const float*)d_in[8];  const float* bv = (const float*)d_in[9];
    const float* Wo = (const float*)d_in[10]; const float* bo = (const float*)d_in[11];
    float* out = (float*)d_out;

    const size_t tsz = (size_t)B_ * S_ * D_;   // 4M elems
    const size_t wsz = (size_t)D_ * D_;        // 1M elems
    __bf16* qx = (__bf16*)d_ws;                // staged bf16 inputs
    __bf16* kx = qx + tsz;
    __bf16* vx = kx + tsz;
    __bf16* wq = vx + tsz;                     // staged bf16 weights
    __bf16* wk = wq + wsz;
    __bf16* wv = wk + wsz;
    __bf16* wo = wv + wsz;
    __bf16* qb = wo + wsz;                     // projected Q/K/V (bf16)
    __bf16* kb = qb + tsz;
    __bf16* vb = kb + tsz;
    __bf16* xb = qx;                           // context reuses dead qx slab

    const int M = B_ * S_;                     // 4096
    const int cthr = 256, celt = 8 * cthr;
    conv_f32_bf16<<<dim3(tsz / celt), cthr, 0, stream>>>(query, qx, (int)tsz);
    conv_f32_bf16<<<dim3(tsz / celt), cthr, 0, stream>>>(key,   kx, (int)tsz);
    conv_f32_bf16<<<dim3(tsz / celt), cthr, 0, stream>>>(value, vx, (int)tsz);
    conv_f32_bf16<<<dim3(wsz / celt), cthr, 0, stream>>>(Wq, wq, (int)wsz);
    conv_f32_bf16<<<dim3(wsz / celt), cthr, 0, stream>>>(Wk, wk, (int)wsz);
    conv_f32_bf16<<<dim3(wsz / celt), cthr, 0, stream>>>(Wv, wv, (int)wsz);
    conv_f32_bf16<<<dim3(wsz / celt), cthr, 0, stream>>>(Wo, wo, (int)wsz);

    dim3 ggrid(D_ / 128, M / 64);              // (8, 64)
    gemm_bias_wmma<__bf16><<<ggrid, 256, 0, stream>>>(qx, wq, bq, qb, M, D_, D_);
    gemm_bias_wmma<__bf16><<<ggrid, 256, 0, stream>>>(kx, wk, bk, kb, M, D_, D_);
    gemm_bias_wmma<__bf16><<<ggrid, 256, 0, stream>>>(vx, wv, bv, vb, M, D_, D_);

    attn_fwd<<<dim3(S_ / 128, B_ * H_), 256, 0, stream>>>(qb, kb, vb, xb);

    gemm_bias_wmma<float><<<ggrid, 256, 0, stream>>>(xb, wo, bo, out, M, D_, D_);
}
